// GraphSAGEConv_25305947308734
// MI455X (gfx1250) — matrix operand order repaired and checked by reading, not compiled, not run on hardware
//
#include <hip/hip_runtime.h>

typedef __attribute__((ext_vector_type(16))) __bf16 v16bf;
typedef __attribute__((ext_vector_type(8)))  float  v8f;

#define N_NODES 50000
#define N_EDGES 800000
#define DIM     256
#define BATCH   4096
#define MT      16          // rows per GEMM workgroup (50000 = 3125*16 exactly)
#define LN_EPS  1e-5f

// ---------- bf16 helpers (round-to-nearest-even) ----------
static __device__ __forceinline__ unsigned short f32_to_bf16(float f) {
  unsigned int u = __float_as_uint(f);
  u += 0x7fffu + ((u >> 16) & 1u);
  return (unsigned short)(u >> 16);
}
static __device__ __forceinline__ float bf16_to_f32(unsigned short h) {
  return __uint_as_float(((unsigned int)h) << 16);
}

union ABu { uint4 q[2]; v16bf v; };   // 32 bytes: one 16-element bf16 WMMA operand

// ---------- CSR build ----------
__global__ void k_zero_i32(int* p, int n) {
  int i = blockIdx.x * blockDim.x + threadIdx.x;
  if (i < n) p[i] = 0;
}
__global__ void k_count(const int* __restrict__ dst, int* __restrict__ deg) {
  int i = blockIdx.x * blockDim.x + threadIdx.x;
  if (i < N_EDGES) atomicAdd(&deg[dst[i]], 1);
}
// single-block chunked Hillis-Steele exclusive scan: rowptr[0..N], rowptr[N]=total
__global__ void k_scan(const int* __restrict__ deg, int* __restrict__ rowptr) {
  __shared__ int buf[1024];
  __shared__ int carry;
  const int t = threadIdx.x;
  if (t == 0) carry = 0;
  __syncthreads();
  for (int base = 0; base < N_NODES; base += 1024) {
    int v = (base + t < N_NODES) ? deg[base + t] : 0;
    buf[t] = v;
    __syncthreads();
    for (int off = 1; off < 1024; off <<= 1) {
      int x = (t >= off) ? buf[t - off] : 0;
      __syncthreads();
      buf[t] += x;
      __syncthreads();
    }
    if (base + t < N_NODES) rowptr[base + t] = carry + (buf[t] - v);
    __syncthreads();
    if (t == 1023) carry += buf[1023];
    __syncthreads();
  }
  if (t == 0) rowptr[N_NODES] = carry;
}
__global__ void k_inv(const int* __restrict__ deg, float* __restrict__ inv) {
  int i = blockIdx.x * blockDim.x + threadIdx.x;
  if (i < N_NODES) inv[i] = 1.0f / ((float)deg[i] + 1.0f);
}
__global__ void k_copy_i32(const int* __restrict__ a, int* __restrict__ b, int n) {
  int i = blockIdx.x * blockDim.x + threadIdx.x;
  if (i < n) b[i] = a[i];
}
__global__ void k_fill(const int* __restrict__ src, const int* __restrict__ dst,
                       int* __restrict__ cursor, int* __restrict__ colsrc) {
  int i = blockIdx.x * blockDim.x + threadIdx.x;
  if (i < N_EDGES) {
    int pos = atomicAdd(&cursor[dst[i]], 1);
    colsrc[pos] = src[i];
  }
}

// ---------- split W (f32) -> bf16 hi/lo (same [o][k] row-major layout as Ws) ----------
__global__ void k_wsplit(const float* __restrict__ W, unsigned short* __restrict__ Whi,
                         unsigned short* __restrict__ Wlo, int n) {
  int i = blockIdx.x * blockDim.x + threadIdx.x;
  if (i < n) {
    float w = W[i];
    unsigned short hi = f32_to_bf16(w);
    Whi[i] = hi;
    Wlo[i] = f32_to_bf16(w - bf16_to_f32(hi));
  }
}

// ---------- CSR gather aggregation: one wave per node, lane owns 8 channels ----------
__global__ __launch_bounds__(256) void k_aggregate(
    const float* __restrict__ feats, float* __restrict__ agg,
    const int* __restrict__ rowptr, const int* __restrict__ colsrc) {
  const int node = blockIdx.x * 8 + (threadIdx.x >> 5);
  if (node >= N_NODES) return;
  const int lane = threadIdx.x & 31;
  const int beg = rowptr[node], end = rowptr[node + 1];
  float4 a0 = {0.f, 0.f, 0.f, 0.f}, a1 = {0.f, 0.f, 0.f, 0.f};
  for (int e = beg; e < end; ++e) {
    const int s = colsrc[e];
    const float4* p = (const float4*)(feats + (size_t)s * DIM + lane * 8);
    float4 x0 = p[0], x1 = p[1];
    a0.x += x0.x; a0.y += x0.y; a0.z += x0.z; a0.w += x0.w;
    a1.x += x1.x; a1.y += x1.y; a1.z += x1.z; a1.w += x1.w;
  }
  float4* q = (float4*)(agg + (size_t)node * DIM + lane * 8);
  q[0] = a0; q[1] = a1;
}

// ---------- fused: x=(agg+self)*inv ; h=x@W^T+b (bf16x3 WMMA) ; LN ; ELU -> feats ----------
__global__ __launch_bounds__(256) void k_gemm_ln_elu(
    const float* __restrict__ agg, float* __restrict__ feats,
    const float* __restrict__ inv,
    const unsigned short* __restrict__ Whi, const unsigned short* __restrict__ Wlo,
    const float* __restrict__ bias, const float* __restrict__ gamma,
    const float* __restrict__ beta) {
  __shared__ unsigned short sXhi[MT * DIM];
  __shared__ unsigned short sXlo[MT * DIM];
  __shared__ float sH[MT * DIM];
  __shared__ float sPart[2][MT][16];
  __shared__ float sMu[MT];
  __shared__ float sRs[MT];

  const int tid = threadIdx.x;
  const int m0 = blockIdx.x * MT;

  // stage tile: x = (agg + self) * inv, split into bf16 hi/lo in LDS
  for (int i = tid; i < MT * DIM; i += 256) {
    const int r = i >> 8;
    const int c = i & (DIM - 1);
    const size_t g = (size_t)(m0 + r) * DIM + c;
    const float x = (agg[g] + feats[g]) * inv[m0 + r];
    const unsigned short hi = f32_to_bf16(x);
    sXhi[i] = hi;
    sXlo[i] = f32_to_bf16(x - bf16_to_f32(hi));
  }
  __syncthreads();

  const int lane = tid & 31;
  const int wave = tid >> 5;      // 8 waves, each owns 2 N-tiles of 16 columns
  const int mrow = lane & 15;
  const int half = lane >> 4;

  v8f acc[2] = {{0,0,0,0,0,0,0,0}, {0,0,0,0,0,0,0,0}};

#pragma unroll
  for (int kt = 0; kt < DIM / 32; ++kt) {
    const int kb = kt * 32;
    // A operand (16x32 bf16): lane<16 -> K{0..15}; lane>=16 -> K{16..31}
    ABu ahi, alo;
    const int aoff = mrow * DIM + kb + half * 16;
    ahi.q[0] = *(const uint4*)(sXhi + aoff);
    ahi.q[1] = *(const uint4*)(sXhi + aoff + 8);
    alo.q[0] = *(const uint4*)(sXlo + aoff);
    alo.q[1] = *(const uint4*)(sXlo + aoff + 8);
#pragma unroll
    for (int s = 0; s < 2; ++s) {
      const int nt = wave * 2 + s;
      // B operand (32x16 bf16): lane holds 16 consecutive K of column o = contiguous W row chunk
      const int o = nt * 16 + mrow;
      const int wb = o * DIM + kb + half * 16;
      const v16bf bhi = *(const v16bf*)(Whi + wb);
      const v16bf blo = *(const v16bf*)(Wlo + wb);
      acc[s] = __builtin_amdgcn_wmma_f32_16x16x32_bf16(false, ahi.v, false, bhi,
                                                       (short)0, acc[s], false, false);
      acc[s] = __builtin_amdgcn_wmma_f32_16x16x32_bf16(false, ahi.v, false, blo,
                                                       (short)0, acc[s], false, false);
      acc[s] = __builtin_amdgcn_wmma_f32_16x16x32_bf16(false, alo.v, false, bhi,
                                                       (short)0, acc[s], false, false);
    }
  }

  // spill D tiles (+bias) into LDS; D layout: VGPR j -> row j + 8*half, col = nt*16 + (lane&15)
#pragma unroll
  for (int s = 0; s < 2; ++s) {
    const int col = (wave * 2 + s) * 16 + mrow;
    const float b = bias[col];
#pragma unroll
    for (int j = 0; j < 8; ++j)
      sH[(half * 8 + j) * DIM + col] = acc[s][j] + b;
  }
  __syncthreads();

  // LayerNorm stats: 16 threads per row, each sums 16 elements
  {
    const int r = tid >> 4, seg = tid & 15;
    float s = 0.f, ss = 0.f;
    for (int i = 0; i < 16; ++i) {
      const float v = sH[r * DIM + seg * 16 + i];
      s += v; ss += v * v;
    }
    sPart[0][r][seg] = s;
    sPart[1][r][seg] = ss;
  }
  __syncthreads();
  if (tid < MT) {
    float s = 0.f, ss = 0.f;
    for (int i = 0; i < 16; ++i) { s += sPart[0][tid][i]; ss += sPart[1][tid][i]; }
    const float mu = s * (1.0f / DIM);
    const float var = ss * (1.0f / DIM) - mu * mu;
    sMu[tid] = mu;
    sRs[tid] = rsqrtf(var + LN_EPS);
  }
  __syncthreads();
  // apply LN + ELU, write feats in place (rows are block-exclusive)
  {
    const int r = tid >> 4, seg = tid & 15;
    const float mu = sMu[r], rs = sRs[r];
    for (int i = 0; i < 16; ++i) {
      const int c = seg * 16 + i;
      float v = (sH[r * DIM + c] - mu) * rs * gamma[c] + beta[c];
      v = (v > 0.f) ? v : (__expf(v) - 1.0f);
      feats[(size_t)(m0 + r) * DIM + c] = v;
    }
  }
}

// ---------- final gather ----------
__global__ void k_gather(const float* __restrict__ feats, const int* __restrict__ index,
                         float* __restrict__ out) {
  const int b = blockIdx.x, t = threadIdx.x;
  out[(size_t)b * DIM + t] = feats[(size_t)index[b] * DIM + t];
}

extern "C" void kernel_launch(void* const* d_in, const int* in_sizes, int n_in,
                              void* d_out, int out_size, void* d_ws, size_t ws_size,
                              hipStream_t stream) {
  (void)in_sizes; (void)n_in; (void)out_size; (void)ws_size;
  const float* embedding = (const float*)d_in[0];
  const float* Ws        = (const float*)d_in[1];
  const float* bs        = (const float*)d_in[2];
  const float* gammas    = (const float*)d_in[3];
  const float* betas     = (const float*)d_in[4];
  const int*   src       = (const int*)d_in[5];
  const int*   dst       = (const int*)d_in[6];
  const int*   index     = (const int*)d_in[7];
  float* out = (float*)d_out;

  // workspace carving (256B aligned): ~107 MB total
  char* p = (char*)d_ws;
  auto carve = [&](size_t bytes) -> char* {
    char* r = p;
    p += (bytes + 255) & ~(size_t)255;
    return r;
  };
  float* feats  = (float*)carve((size_t)N_NODES * DIM * 4);
  float* agg    = (float*)carve((size_t)N_NODES * DIM * 4);
  float* inv    = (float*)carve((size_t)N_NODES * 4);
  int*   deg    = (int*)carve((size_t)N_NODES * 4);
  int*   rowptr = (int*)carve((size_t)(N_NODES + 1) * 4);
  int*   cursor = (int*)carve((size_t)N_NODES * 4);
  int*   colsrc = (int*)carve((size_t)N_EDGES * 4);
  unsigned short* Whi = (unsigned short*)carve((size_t)2 * DIM * DIM * 2);
  unsigned short* Wlo = (unsigned short*)carve((size_t)2 * DIM * DIM * 2);

  hipMemcpyAsync(feats, embedding, (size_t)N_NODES * DIM * 4,
                 hipMemcpyDeviceToDevice, stream);

  // CSR build (degree is layer-invariant)
  k_zero_i32<<<(N_NODES + 255) / 256, 256, 0, stream>>>(deg, N_NODES);
  k_count<<<(N_EDGES + 255) / 256, 256, 0, stream>>>(dst, deg);
  k_scan<<<1, 1024, 0, stream>>>(deg, rowptr);
  k_inv<<<(N_NODES + 255) / 256, 256, 0, stream>>>(deg, inv);
  k_copy_i32<<<(N_NODES + 255) / 256, 256, 0, stream>>>(rowptr, cursor, N_NODES);
  k_fill<<<(N_EDGES + 255) / 256, 256, 0, stream>>>(src, dst, cursor, colsrc);
  k_wsplit<<<(2 * DIM * DIM + 255) / 256, 256, 0, stream>>>(Ws, Whi, Wlo, 2 * DIM * DIM);

  for (int layer = 0; layer < 2; ++layer) {
    k_aggregate<<<(N_NODES + 7) / 8, 256, 0, stream>>>(feats, agg, rowptr, colsrc);
    k_gemm_ln_elu<<<N_NODES / MT, 256, 0, stream>>>(
        agg, feats, inv,
        Whi + (size_t)layer * DIM * DIM, Wlo + (size_t)layer * DIM * DIM,
        bs + layer * DIM, gammas + layer * DIM, betas + layer * DIM);
  }
  k_gather<<<BATCH, 256, 0, stream>>>(feats, index, out);
}
